// inception_block_dcn_58815282152120
// MI455X (gfx1250) — matrix operand order repaired and checked by reading, not compiled, not run on hardware
//
#include <hip/hip_runtime.h>
#include <hip/hip_bf16.h>
#include <math.h>

// ---------------------------------------------------------------------------
// Problem constants (from the reference): B=2, C=32, H=W=256, f32 everywhere.
// ---------------------------------------------------------------------------
#define NB     2
#define CIN    32
#define IMG_H  256
#define IMG_W  256
#define IMG_HW (IMG_H * IMG_W)
#define TILE_W 16
#define EPSV   1e-5f

typedef __attribute__((ext_vector_type(16))) __bf16 v16bf;
typedef __attribute__((ext_vector_type(8)))  float  v8f;

__device__ __forceinline__ v8f wmma_bf16(v16bf a, v16bf b, v8f c) {
  // D = A(16x32 bf16) * B(32x16 bf16) + C(f32), one wave.
  return __builtin_amdgcn_wmma_f32_16x16x32_bf16(
      /*neg_a=*/false, a, /*neg_b=*/false, b,
      /*c_mod=*/(short)0, c, /*reuse_a=*/false, /*reuse_b=*/false);
}

__device__ __forceinline__ int iclampi(int v, int lo, int hi) {
  return v < lo ? lo : (v > hi ? hi : v);
}

// ---------------------------------------------------------------------------
// Tensor Data Mover staging (gfx1250). Guarded: falls back to plain loads if
// the builtin is absent. Descriptor per CDNA5 ISA:
//   group0: [1:0]count=1 | [63:32]lds_addr | [120:64]global_addr | [127:126]=2
//   group1: data_size=2(4B), tensor_dim0/1, tile_dim0/1/2, dim0/dim1 strides
//   group2: tensor_dim2 (large), rest 0;  group3: zeros
// Arity differs per toolchain: 5 args (clang-22 / ROCm 7.2), 6 args (clang-23).
// ---------------------------------------------------------------------------
#if __has_builtin(__builtin_amdgcn_tensor_load_to_lds) && \
    __has_builtin(__builtin_amdgcn_s_wait_tensorcnt) &&   \
    __has_builtin(__builtin_amdgcn_readfirstlane)
#define HAVE_TDM 1
typedef __attribute__((ext_vector_type(4))) unsigned int u32x4;
typedef __attribute__((ext_vector_type(4))) int          i32x4;
typedef __attribute__((ext_vector_type(8))) int          i32x8;

__device__ __forceinline__ void tdm_load_tile_f32(
    const float* gsrc, void* lds_dst,
    int tile_cols, int tile_rows, int tile_ch,
    int row_stride, int ch_stride) {
  const unsigned long long ga = (unsigned long long)(uintptr_t)gsrc;
  const unsigned int lds = (unsigned int)(uintptr_t)lds_dst;  // flat[31:0] = LDS addr
  const unsigned int td0 = 1u << 20, td1 = 1u << 20;          // generous extents

  u32x4 g0;
  g0.x = 1u;                                                   // count=1, user D#
  g0.y = (unsigned)__builtin_amdgcn_readfirstlane((int)lds);
  g0.z = (unsigned)__builtin_amdgcn_readfirstlane((int)(unsigned int)ga);
  g0.w = (unsigned)__builtin_amdgcn_readfirstlane(
             (int)(unsigned int)(((ga >> 32) & 0x01FFFFFFull) | (2u << 30)));

  i32x8 g1;
  g1[0] = 0x00020000;                                          // data_size=4B
  g1[1] = (int)((td0 & 0xFFFFu) << 16);                        // dim0[15:0]<<16
  g1[2] = (int)(((td0 >> 16) & 0xFFFFu) | ((td1 & 0xFFFFu) << 16));
  g1[3] = (int)(((td1 >> 16) & 0xFFFFu) | ((unsigned)tile_cols << 16));
  g1[4] = (int)(((unsigned)tile_rows & 0xFFFFu) | ((unsigned)tile_ch << 16));
  g1[5] = row_stride;                                          // dim0_stride[31:0]
  g1[6] = (int)(((unsigned)ch_stride & 0xFFFFu) << 16);        // dim1_stride[15:0]
  g1[7] = (int)((unsigned)ch_stride >> 16);                    // dim1_stride[47:16]

  i32x4 g2 = {(int)(1u << 20), 0, 0, 0};                       // tensor_dim2 large
  i32x4 g3 = {0, 0, 0, 0};
#if defined(__clang_major__) && (__clang_major__ >= 23)
  i32x8 z8 = {0, 0, 0, 0, 0, 0, 0, 0};
  __builtin_amdgcn_tensor_load_to_lds(g0, g1, g2, g3, z8, 0);
#else
  __builtin_amdgcn_tensor_load_to_lds(g0, g1, g2, g3, 0);
#endif
}
#else
#define HAVE_TDM 0
#endif

// ---------------------------------------------------------------------------
// A 16x32 layout (per lane element i -> K), contiguous to match B:
//   lanes 0-15  (M=lane):    elems 0..15 -> K=0..15
//   lanes 16-31 (M=lane-16): elems 0..15 -> K=16..31
// ---------------------------------------------------------------------------
__device__ __forceinline__ v16bf gather_wA(const float* __restrict__ wt, int mtile,
                                           int lane, int t, int ntaps,
                                           int cin_block, int cin_total, int mout) {
  int m = (lane & 15) + mtile * 16;
  if (m >= mout) m = 0;                      // padded rows (offset conv 27->32)
  const int koff = (lane < 16) ? 0 : 16;
  v16bf r;
#pragma unroll
  for (int i = 0; i < 16; ++i) {
    const int k   = i + koff;
    const int cin = cin_block * 32 + k;
    r[i] = (__bf16)wt[((size_t)m * cin_total + cin) * ntaps + t];
  }
  return r;
}

// Prepacked fragment fetch: one 32-byte contiguous load per lane.
__device__ __forceinline__ v16bf load_fragA(const __bf16* __restrict__ wf,
                                            int frag, int lane) {
  return *(const v16bf*)(wf + ((size_t)frag * 32 + lane) * 16);
}

// ---------------------------------------------------------------------------
// B-fragment direct path (no tap reuse along W):
//   lanes 0-15  (N=lane): elem i -> K=i; lanes 16-31: elem i -> K=16+i
// ---------------------------------------------------------------------------
__device__ __forceinline__ v16bf load_actB(const float* __restrict__ xin, int bimg,
                                           int h, int w0, int lane, int dy, int dx) {
  const int n     = lane & 15;
  const int cbase = (lane < 16) ? 0 : 16;
  const int hh    = h + dy;
  const int ww    = w0 + n + dx;
  const bool inb  = (hh >= 0) && (hh < IMG_H) && (ww >= 0) && (ww < IMG_W);
  const float* base = xin + (size_t)bimg * CIN * IMG_HW + (size_t)hh * IMG_W + ww;
  v16bf r;
#pragma unroll
  for (int i = 0; i < 16; ++i) {
    float v = inb ? base[(size_t)(cbase + i) * IMG_HW] : 0.0f;
    r[i] = (__bf16)v;
  }
  return r;
}

// ---------------------------------------------------------------------------
// Prep kernel 1: pack one weight matrix into WMMA-ready bf16 A-fragments.
// grid.x = ntaps*2 fragments, block = 32. cin_mode: "taps" = 32ch K-blocks.
// ---------------------------------------------------------------------------
__global__ __launch_bounds__(32) void wprep_kernel(
    const float* __restrict__ wt, int ntaps, int cin_total, int mout,
    int cin_mode, __bf16* __restrict__ dst) {
  const int frag  = blockIdx.x;
  const int t     = frag >> 1;
  const int mtile = frag & 1;
  const int lane  = threadIdx.x & 31;
  v16bf a = cin_mode
                ? gather_wA(wt, mtile, lane, 0, 1, t, cin_total, mout)
                : gather_wA(wt, mtile, lane, t, ntaps, 0, cin_total, mout);
  *(v16bf*)(dst + ((size_t)frag * 32 + lane) * 16) = a;
}

// ---------------------------------------------------------------------------
// Prep kernel 2: fold bias + inference BN: y = acc*sc[m] + sh[m].
// ---------------------------------------------------------------------------
__global__ __launch_bounds__(32) void bnprep_kernel(
    const float* __restrict__ bias, const float* __restrict__ g,
    const float* __restrict__ b, const float* __restrict__ m,
    const float* __restrict__ v, int mout,
    float* __restrict__ sc, float* __restrict__ sh) {
  const int c = threadIdx.x;
  if (c < mout) {
    if (g) {
      const float s = g[c] * rsqrtf(v[c] + EPSV);
      sc[c] = s;
      sh[c] = (bias[c] - m[c]) * s + b[c];
    } else {
      sc[c] = 1.0f;
      sh[c] = bias[c];
    }
  }
}

// ---------------------------------------------------------------------------
// Generic fused conv. One wave owns a MOUTx16-pixel tile (2 WMMA accumulators).
// KW>1: the 32ch x KH x (16+KW-1) window is staged in LDS as bf16
// [row][col][channel]; interior tiles arrive via TDM (raw f32 [ch][row][col],
// then a transpose/convert pass), border tiles via plain coalesced loads.
// ---------------------------------------------------------------------------
template <int KH, int KW, int WPB, int MOUT, bool DO_RELU, bool SIG_TAIL>
__global__ __launch_bounds__(WPB * 32) void conv_wmma_kernel(
    const float* __restrict__ xin, const __bf16* __restrict__ wf,
    const float* __restrict__ sc, const float* __restrict__ sh,
    float* __restrict__ out) {
  __builtin_prefetch(wf, 0, 1);
  constexpr int  CW      = 16 + KW - 1;          // staged columns
  constexpr bool USE_LDS = (KW > 1);             // tap reuse exists only along W
  __shared__ __align__(32) __bf16 tile_s[USE_LDS ? WPB * KH * CW * 32 : 32];

  const int lane  = threadIdx.x & 31;
  const int wslot = threadIdx.x >> 5;
  const int wave  = blockIdx.x * (blockDim.x >> 5) + wslot;
  const int p0    = wave * TILE_W;
  const int bimg  = p0 / IMG_HW;
  const int rem   = p0 % IMG_HW;
  const int h     = rem / IMG_W;
  const int w0    = rem % IMG_W;                 // multiple of 16, row-aligned
  const int n     = lane & 15;
  const int cbase = (lane < 16) ? 0 : 16;

  __bf16* tile = &tile_s[(size_t)wslot * KH * CW * 32];
  if (USE_LDS) {
    constexpr int ITERS = KH * CW;               // (KH*CW*32)/32 lanes
#if HAVE_TDM
    __shared__ __align__(16) float raw_s[USE_LDS ? WPB * KH * CW * 32 : 8];
    float* raw = &raw_s[(size_t)wslot * KH * CW * 32];
    const bool interior = (h - KH / 2 >= 0) && (h + KH / 2 < IMG_H) &&
                          (w0 - KW / 2 >= 0) && (w0 + 15 + KW / 2 < IMG_W);
    if (interior) {                              // wave-uniform branch
      const float* gsrc = xin + (size_t)bimg * CIN * IMG_HW +
                          (size_t)(h - KH / 2) * IMG_W + (w0 - KW / 2);
      tdm_load_tile_f32(gsrc, raw, CW, KH, CIN, IMG_W, IMG_HW);
      __builtin_amdgcn_s_wait_tensorcnt(0);
      asm volatile("" ::: "memory");             // order LDS reads after TDM
#pragma unroll 1
      for (int it = 0; it < ITERS; ++it) {
        const int e   = it * 32 + lane;
        const int col = e % CW;
        const int rc  = e / CW;
        const int ch  = rc & 31;
        const int r   = rc >> 5;
        tile[((size_t)r * CW + col) * 32 + ch] =
            (__bf16)raw[((size_t)ch * KH + r) * CW + col];
      }
    } else
#endif
    {
#pragma unroll 1
      for (int it = 0; it < ITERS; ++it) {
        const int e   = it * 32 + lane;
        const int col = e % CW;
        const int rc  = e / CW;
        const int ch  = rc & 31;
        const int r   = rc >> 5;
        const int hh  = h + r - KH / 2;
        const int ww  = w0 + col - KW / 2;
        const bool inb = (hh >= 0) && (hh < IMG_H) && (ww >= 0) && (ww < IMG_W);
        const float v = inb
            ? xin[((size_t)bimg * CIN + ch) * IMG_HW + (size_t)hh * IMG_W + ww]
            : 0.0f;
        tile[((size_t)r * CW + col) * 32 + ch] = (__bf16)v;
      }
    }
  }

  v8f acc0 = {}; v8f acc1 = {};
#pragma unroll
  for (int t = 0; t < KH * KW; ++t) {
    v16bf bf;
    if (USE_LDS) {
      const int r   = t / KW;
      const int col = n + t % KW;                // window col for this pixel/tap
      bf = *(const v16bf*)(tile + ((size_t)r * CW + col) * 32 + cbase);
    } else {
      bf = load_actB(xin, bimg, h, w0, lane, t / KW - KH / 2, t % KW - KW / 2);
    }
    v16bf a0 = load_fragA(wf, t * 2 + 0, lane);
    v16bf a1 = load_fragA(wf, t * 2 + 1, lane);
    acc0 = wmma_bf16(a0, bf, acc0);
    acc1 = wmma_bf16(a1, bf, acc1);
  }

  // C/D layout: VGPR r, lanes 0-15 -> M=r, lanes 16-31 -> M=r+8, N=lane%16.
  const int moff = (lane < 16) ? 0 : 8;
  const size_t pix = (size_t)h * IMG_W + w0 + n;
#pragma unroll
  for (int half = 0; half < 2; ++half) {
    v8f acc = half ? acc1 : acc0;
#pragma unroll
    for (int r = 0; r < 8; ++r) {
      const int m = half * 16 + moff + r;
      if (m >= MOUT) continue;
      float v = acc[r] * sc[m] + sh[m];
      if (DO_RELU) v = fmaxf(v, 0.0f);
      if (SIG_TAIL && m >= 18) v = 1.0f / (1.0f + __expf(-v));  // mask channels
      out[((size_t)bimg * MOUT + m) * IMG_HW + pix] = v;
    }
  }
}

// ---------------------------------------------------------------------------
// Modulated deformable conv (DCNv2 3x3): VALU bilinear gather (zero pad +
// mask folded into the 4 corner weights) -> bf16 B-fragment -> WMMA.
// ---------------------------------------------------------------------------
__global__ __launch_bounds__(256) void dcn_wmma_kernel(
    const float* __restrict__ y, const float* __restrict__ om,
    const __bf16* __restrict__ wf, float* __restrict__ out) {
  const int lane = threadIdx.x & 31;
  const int wave = blockIdx.x * (blockDim.x >> 5) + (threadIdx.x >> 5);
  const int p0   = wave * TILE_W;
  const int bimg = p0 / IMG_HW;
  const int rem  = p0 % IMG_HW;
  const int h    = rem / IMG_W;
  const int w0   = rem % IMG_W;
  const int n     = lane & 15;
  const int cbase = (lane < 16) ? 0 : 16;
  const size_t pix = (size_t)h * IMG_W + w0 + n;
  const float* ombase = om + (size_t)bimg * 27 * IMG_HW + pix;
  const float* ybase  = y  + (size_t)bimg * CIN * IMG_HW;

  v8f acc0 = {}; v8f acc1 = {};
#pragma unroll 1
  for (int s = 0; s < 9; ++s) {
    const float dyv = ombase[(size_t)(2 * s + 0) * IMG_HW];
    const float dxv = ombase[(size_t)(2 * s + 1) * IMG_HW];
    const float msk = ombase[(size_t)(18 + s) * IMG_HW];
    const float py = (float)h + (float)(s / 3 - 1) + dyv;
    const float px = (float)(w0 + n) + (float)(s % 3 - 1) + dxv;
    const float fy0 = floorf(py), fx0 = floorf(px);
    const float fy = py - fy0, fx = px - fx0;
    const int yi = (int)fy0, xi = (int)fx0;
    const bool vy0 = (yi     >= 0) && (yi     < IMG_H);
    const bool vy1 = (yi + 1 >= 0) && (yi + 1 < IMG_H);
    const bool vx0 = (xi     >= 0) && (xi     < IMG_W);
    const bool vx1 = (xi + 1 >= 0) && (xi + 1 < IMG_W);
    const int yc0 = iclampi(yi, 0, IMG_H - 1), yc1 = iclampi(yi + 1, 0, IMG_H - 1);
    const int xc0 = iclampi(xi, 0, IMG_W - 1), xc1 = iclampi(xi + 1, 0, IMG_W - 1);
    const float w00 = (1.f - fy) * (1.f - fx) * ((vy0 && vx0) ? msk : 0.f);
    const float w01 = (1.f - fy) * fx         * ((vy0 && vx1) ? msk : 0.f);
    const float w10 = fy * (1.f - fx)         * ((vy1 && vx0) ? msk : 0.f);
    const float w11 = fy * fx                 * ((vy1 && vx1) ? msk : 0.f);
    const int i00 = yc0 * IMG_W + xc0, i01 = yc0 * IMG_W + xc1;
    const int i10 = yc1 * IMG_W + xc0, i11 = yc1 * IMG_W + xc1;
    v16bf bf;
#pragma unroll
    for (int i = 0; i < 16; ++i) {
      const float* ch = ybase + (size_t)(cbase + i) * IMG_HW;
      const float v = ch[i00] * w00 + ch[i01] * w01 + ch[i10] * w10 + ch[i11] * w11;
      bf[i] = (__bf16)v;
    }
    v16bf a0 = load_fragA(wf, s * 2 + 0, lane);
    v16bf a1 = load_fragA(wf, s * 2 + 1, lane);
    acc0 = wmma_bf16(a0, bf, acc0);
    acc1 = wmma_bf16(a1, bf, acc1);
  }

  const int moff = (lane < 16) ? 0 : 8;
#pragma unroll
  for (int half = 0; half < 2; ++half) {
    v8f acc = half ? acc1 : acc0;
#pragma unroll
    for (int r = 0; r < 8; ++r) {
      const int m = half * 16 + moff + r;
      out[((size_t)bimg * 32 + m) * IMG_HW + pix] = acc[r];  // DCN has no bias
    }
  }
}

// ---------------------------------------------------------------------------
// Concat(4x32=128) -> 1x1 conv -> +bias -> ReLU; concat never materialized.
// ---------------------------------------------------------------------------
__global__ __launch_bounds__(256) void cat_conv_relu_kernel(
    const float* __restrict__ d0, const float* __restrict__ d1,
    const float* __restrict__ d2, const float* __restrict__ d3,
    const __bf16* __restrict__ wf,
    const float* __restrict__ sc, const float* __restrict__ sh,
    float* __restrict__ out) {
  const int lane = threadIdx.x & 31;
  const int wave = blockIdx.x * (blockDim.x >> 5) + (threadIdx.x >> 5);
  const int p0   = wave * TILE_W;
  const int bimg = p0 / IMG_HW;
  const int rem  = p0 % IMG_HW;
  const int h    = rem / IMG_W;
  const int w0   = rem % IMG_W;

  const float* ins[4] = {d0, d1, d2, d3};
  v8f acc0 = {}; v8f acc1 = {};
#pragma unroll
  for (int kb = 0; kb < 4; ++kb) {
    v16bf bf = load_actB(ins[kb], bimg, h, w0, lane, 0, 0);
    v16bf a0 = load_fragA(wf, kb * 2 + 0, lane);
    v16bf a1 = load_fragA(wf, kb * 2 + 1, lane);
    acc0 = wmma_bf16(a0, bf, acc0);
    acc1 = wmma_bf16(a1, bf, acc1);
  }
  const int n    = lane & 15;
  const int moff = (lane < 16) ? 0 : 8;
  const size_t pix = (size_t)h * IMG_W + w0 + n;
#pragma unroll
  for (int half = 0; half < 2; ++half) {
    v8f acc = half ? acc1 : acc0;
#pragma unroll
    for (int r = 0; r < 8; ++r) {
      const int m = half * 16 + moff + r;
      out[((size_t)bimg * 32 + m) * IMG_HW + pix] =
          fmaxf(acc[r] * sc[m] + sh[m], 0.0f);
    }
  }
}

// ---------------------------------------------------------------------------
// Host-side orchestration. Input flattening = dict insertion order, depth-first:
//   0:x | b1: 1:c1_w 2:c1_b 3..6:bn1(g,b,m,v) 7:off_w 8:off_b 9:dcn_w
//   b2 @10 (c1_w,c1_b,bn1*4,c2_w,c2_b,bn2*4,off_w,off_b,dcn_w), b3 @25, b4 @40
//   55:cat_w 56:cat_b 57:ygx_w 58:ygx_b 59..62:ygx_bn(g,b,m,v)
// ---------------------------------------------------------------------------
extern "C" void kernel_launch(void* const* d_in, const int* in_sizes, int n_in,
                              void* d_out, int out_size, void* d_ws, size_t ws_size,
                              hipStream_t stream) {
  (void)in_sizes; (void)n_in; (void)out_size; (void)ws_size;
  auto in = [&](int i) { return (const float*)d_in[i]; };
  const float* x = in(0);
  float* ws = (float*)d_ws;

  const size_t ACT = (size_t)NB * 32 * IMG_HW;  // 4,194,304 floats (16 MB)
  const size_t OFF = (size_t)NB * 27 * IMG_HW;  // 3,538,944 floats
  float* tmp  = ws;                // c1 output for 2-conv branches; later y0
  float* ybuf = ws + ACT;          // branch activation fed to offset conv + DCN
  float* off  = ws + 2 * ACT;      // offset/mask buffer (reused per branch)
  float* dcn0 = ws + 2 * ACT + OFF;
  float* dcn1 = dcn0 + ACT;
  float* dcn2 = dcn1 + ACT;
  float* dcn3 = dcn2 + ACT;
  float* y0   = tmp;               // safe: branches are done before cat
  float* epi  = dcn3 + ACT;        // 13 layers x {sc[32], sh[32]}
  __bf16* wfb = (__bf16*)(epi + 13 * 64);  // prepacked A-fragments (32B aligned)

  // ---- prep: pack weight fragments (17 weight tensors) -------------------
  const int wt_idx[17]  = {1, 7, 9, 10, 16, 22, 24, 25, 31, 37, 39,
                           40, 46, 52, 54, 55, 57};
  const int wt_taps[17] = {1, 9, 9, 3, 3, 9, 9, 5, 5, 9, 9, 7, 7, 9, 9, 4, 9};
  const int wt_is_off[17] = {0, 1, 0, 0, 0, 1, 0, 0, 0, 1, 0, 0, 0, 1, 0, 0, 0};
  size_t wf_off[17];
  {
    size_t cur = 0;
    for (int l = 0; l < 17; ++l) {
      wf_off[l] = cur;
      const int taps = wt_taps[l];
      wprep_kernel<<<dim3(taps * 2), dim3(32), 0, stream>>>(
          in(wt_idx[l]), taps, (l == 15) ? 128 : 32, wt_is_off[l] ? 27 : 32,
          (l == 15) ? 1 : 0, wfb + cur);
      cur += (size_t)taps * 2 * 32 * 16;
    }
  }

  // ---- prep: per-channel scale/shift epilogues (13 layers) ---------------
  const int ep_bias[13] = {2, 8, 11, 17, 23, 26, 32, 38, 41, 47, 53, 56, 58};
  const int ep_bn[13]   = {3, -1, 12, 18, -1, 27, 33, -1, 42, 48, -1, -1, 59};
  for (int l = 0; l < 13; ++l) {
    const int mout = (l == 1 || l == 4 || l == 7 || l == 10) ? 27 : 32;
    const float* g = ep_bn[l] >= 0 ? in(ep_bn[l] + 0) : nullptr;
    const float* b = ep_bn[l] >= 0 ? in(ep_bn[l] + 1) : nullptr;
    const float* m = ep_bn[l] >= 0 ? in(ep_bn[l] + 2) : nullptr;
    const float* v = ep_bn[l] >= 0 ? in(ep_bn[l] + 3) : nullptr;
    bnprep_kernel<<<dim3(1), dim3(32), 0, stream>>>(
        in(ep_bias[l]), g, b, m, v, mout, epi + l * 64, epi + l * 64 + 32);
  }
  auto SC = [&](int l) { return (const float*)(epi + l * 64); };
  auto SH = [&](int l) { return (const float*)(epi + l * 64 + 32); };

  const int NW = NB * IMG_HW / TILE_W;          // 8192 tile-waves per layer

#define CBR(KH, KW, WPB, SRC, DST, WL, EL)                                     \
  conv_wmma_kernel<KH, KW, WPB, 32, true, false>                               \
      <<<dim3(NW / WPB), dim3(WPB * 32), 0, stream>>>(                         \
          SRC, wfb + wf_off[WL], SC(EL), SH(EL), DST)
#define OFFC(SRC, WL, EL, DST)                                                 \
  conv_wmma_kernel<3, 3, 4, 27, false, true>                                   \
      <<<dim3(NW / 4), dim3(128), 0, stream>>>(                                \
          SRC, wfb + wf_off[WL], SC(EL), SH(EL), DST)
#define DCN(SRC, OM, WL, DST)                                                  \
  dcn_wmma_kernel<<<dim3(NW / 8), dim3(256), 0, stream>>>(                     \
      SRC, OM, wfb + wf_off[WL], DST)

  // branch 1: 1x1
  CBR(1, 1, 8, x, ybuf, 0, 0);
  OFFC(ybuf, 1, 1, off);
  DCN(ybuf, off, 2, dcn0);
  // branch 2: 1x3 + 3x1
  CBR(1, 3, 8, x, tmp, 3, 2);
  CBR(3, 1, 8, tmp, ybuf, 4, 3);
  OFFC(ybuf, 5, 4, off);
  DCN(ybuf, off, 6, dcn1);
  // branch 3: 1x5 + 5x1
  CBR(1, 5, 8, x, tmp, 7, 5);
  CBR(5, 1, 8, tmp, ybuf, 8, 6);
  OFFC(ybuf, 9, 7, off);
  DCN(ybuf, off, 10, dcn2);
  // branch 4: 1x7 + 7x1
  CBR(1, 7, 8, x, tmp, 11, 8);
  CBR(7, 1, 8, tmp, ybuf, 12, 9);
  OFFC(ybuf, 13, 10, off);
  DCN(ybuf, off, 14, dcn3);

  // concat -> 1x1 conv -> ReLU (reference applies no BN here)
  cat_conv_relu_kernel<<<dim3(NW / 8), dim3(256), 0, stream>>>(
      dcn0, dcn1, dcn2, dcn3, wfb + wf_off[15], SC(11), SH(11), y0);
  // ygx: 3x3 conv + bias + BN + ReLU -> final output
  conv_wmma_kernel<3, 3, 4, 32, true, false>
      <<<dim3(NW / 4), dim3(128), 0, stream>>>(
          y0, wfb + wf_off[16], SC(12), SH(12), (float*)d_out);

#undef CBR
#undef OFFC
#undef DCN
}
